// ExpertGQALayer_65189013619204
// MI455X (gfx1250) — compile-verified, matrix operand order checked by reading, not run on hardware
//
#include <hip/hip_runtime.h>
#include <hip/hip_bf16.h>

// ---------------------------------------------------------------------------
// MI455X (gfx1250) transformer layer: RMSNorm -> QKV -> RoPE -> GQA attention
// -> O-proj(+res) -> RMSNorm -> SwiGLU MLP(+res).
// Compute-bound (~515 GFLOP vs ~0.3 GB HBM) => all matmuls via
// v_wmma_f32_16x16x32_bf16 (bf16 in, fp32 accum). Elementwise epilogues fp32.
// Main GEMM: 128x128 block tile, 64x32 wave tile (8 wmma / K-step / wave),
// double-buffered LDS -> one barrier per K-tile.
// ---------------------------------------------------------------------------

typedef __bf16 bf16_t;
typedef __attribute__((ext_vector_type(8)))  __bf16 v8bf;
typedef __attribute__((ext_vector_type(16))) __bf16 v16bf;
typedef __attribute__((ext_vector_type(8)))  float  v8f;

union FragBF { v16bf v; v8bf h[2]; };

constexpr int B_ = 8, S_ = 512, H_ = 2048;
constexpr int NQ_ = 16, NKV_ = 4, HD_ = 128, INTER_ = 8192;
constexpr int G_ = NQ_ / NKV_;
constexpr float EPS_ = 1e-6f;
constexpr float SCALE_ = 0.08838834764831845f;   // 1/sqrt(128)
constexpr float LOG_THETA_ = 11.512925464970229f; // ln(100000)

__device__ __forceinline__ bf16_t f2bf(float f) {
  unsigned u = __builtin_bit_cast(unsigned, f);
  unsigned r = u + 0x7FFFu + ((u >> 16) & 1u);   // round-to-nearest-even
  unsigned short hs = (unsigned short)(r >> 16);
  return __builtin_bit_cast(bf16_t, hs);
}

// ---------------------------------------------------------------------------
// fp32 -> bf16 conversion (weights)
// ---------------------------------------------------------------------------
__global__ __launch_bounds__(256) void cvt_k(const float* __restrict__ in,
                                             bf16_t* __restrict__ out,
                                             long long n) {
  long long i = (long long)blockIdx.x * blockDim.x + threadIdx.x;
  long long st = (long long)gridDim.x * blockDim.x;
  for (; i < n; i += st) out[i] = f2bf(in[i]);
}

// ---------------------------------------------------------------------------
// RMSNorm: row of H_=2048, 256 threads x 8 elems, bf16 out
// ---------------------------------------------------------------------------
__global__ __launch_bounds__(256) void rmsnorm_k(const float* __restrict__ x,
                                                 const float* __restrict__ w,
                                                 bf16_t* __restrict__ out) {
  long long row = blockIdx.x;
  const float* xr = x + row * H_;
  int tid = threadIdx.x;
  float v[8];
  float ss = 0.f;
#pragma unroll
  for (int i = 0; i < 8; ++i) {
    v[i] = xr[tid + 256 * i];
    ss += v[i] * v[i];
  }
  __shared__ float red[256];
  red[tid] = ss;
  __syncthreads();
  for (int o = 128; o > 0; o >>= 1) {
    if (tid < o) red[tid] += red[tid + o];
    __syncthreads();
  }
  float inv = rsqrtf(red[0] * (1.0f / H_) + EPS_);
#pragma unroll
  for (int i = 0; i < 8; ++i)
    out[row * H_ + tid + 256 * i] = f2bf(v[i] * inv * w[tid + 256 * i]);
}

// ---------------------------------------------------------------------------
// Batched WMMA GEMM: C[z] = alpha * A[z] @ W[z/wDiv]^T (+ R)
// A: [M,K] bf16 row-major, W: [N,K] bf16 row-major, C: [M,N] fp32.
// Block tile 128x128, BK=32; 8 waves in 2(M)x4(N), wave tile 64x32
// (4x2 wmma tiles). Double-buffered LDS, one barrier per K-tile.
// ---------------------------------------------------------------------------
constexpr int BM = 128, BN = 128, BK = 32, LK = 40; // LK: padded LDS row (bf16)
constexpr int LBUF = BM * LK;                       // one LDS buffer (elems)

template <int EPI> // 0: C=alpha*acc   1: C=alpha*acc + R
__global__ __launch_bounds__(256) void gemm_k(
    const bf16_t* __restrict__ A, const bf16_t* __restrict__ W,
    float* __restrict__ Cf, const float* __restrict__ R,
    int M, int N, int K,
    long long aStr, long long wStr, long long cStr, int wDiv, float alpha) {
  __shared__ __align__(16) bf16_t sA[2 * LBUF];
  __shared__ __align__(16) bf16_t sW[2 * LBUF];

  int z = blockIdx.z;
  A += (long long)z * aStr;
  W += (long long)(z / wDiv) * wStr;
  long long cOff = (long long)z * cStr;

  int tid = threadIdx.x;
  int lane = tid & 31, wid = tid >> 5;
  int wm = wid & 1, wn = wid >> 1; // 2(M) x 4(N) wave grid
  int mBlk = blockIdx.y * BM, nBlk = blockIdx.x * BN;

  v8f acc[4][2] = {};

  // tile loading map: 512 chunks of 8 bf16 per operand, 2 per thread
  int r0 = tid >> 2, cc = tid & 3;
  int r1 = r0 + 64;
  const bf16_t* pa0 = &A[(long long)(mBlk + r0) * K + cc * 8];
  const bf16_t* pa1 = &A[(long long)(mBlk + r1) * K + cc * 8];
  const bf16_t* pw0 = &W[(long long)(nBlk + r0) * K + cc * 8];
  const bf16_t* pw1 = &W[(long long)(nBlk + r1) * K + cc * 8];
  int sOff0 = r0 * LK + cc * 8;
  int sOff1 = r1 * LK + cc * 8;

  int T = K / BK;
  v8bf ra0 = *(const v8bf*)pa0;
  v8bf ra1 = *(const v8bf*)pa1;
  v8bf rw0 = *(const v8bf*)pw0;
  v8bf rw1 = *(const v8bf*)pw1;
  *(v8bf*)&sA[sOff0] = ra0;
  *(v8bf*)&sA[sOff1] = ra1;
  *(v8bf*)&sW[sOff0] = rw0;
  *(v8bf*)&sW[sOff1] = rw1;

  int rr = lane & 15, half = lane >> 4;
  int k0 = half * 8, k1 = 16 + half * 8;
  int buf = 0;

  for (int t = 0; t < T; ++t) {
    __syncthreads();
    if (t + 1 < T) {
      int kt = (t + 1) * BK;
      ra0 = *(const v8bf*)(pa0 + kt);
      ra1 = *(const v8bf*)(pa1 + kt);
      rw0 = *(const v8bf*)(pw0 + kt);
      rw1 = *(const v8bf*)(pw1 + kt);
      if (t + 2 < T) { // global_prefetch_b8 for K-tile t+2
        __builtin_prefetch(pa0 + kt + BK, 0, 1);
        __builtin_prefetch(pw0 + kt + BK, 0, 1);
      }
    }
    const bf16_t* cA = &sA[buf * LBUF];
    const bf16_t* cW = &sW[buf * LBUF];
    FragBF a[4], b[2];
#pragma unroll
    for (int mi = 0; mi < 4; ++mi) {
      const bf16_t* p = &cA[(wm * 64 + mi * 16 + rr) * LK];
      a[mi].h[0] = *(const v8bf*)(p + k0);
      a[mi].h[1] = *(const v8bf*)(p + k1);
    }
#pragma unroll
    for (int ni = 0; ni < 2; ++ni) {
      const bf16_t* p = &cW[(wn * 32 + ni * 16 + rr) * LK];
      b[ni].h[0] = *(const v8bf*)(p + k0);
      b[ni].h[1] = *(const v8bf*)(p + k1);
    }
#pragma unroll
    for (int mi = 0; mi < 4; ++mi)
#pragma unroll
      for (int ni = 0; ni < 2; ++ni)
        acc[mi][ni] = __builtin_amdgcn_wmma_f32_16x16x32_bf16(
            false, a[mi].v, false, b[ni].v, (short)0, acc[mi][ni], false, false);

    if (t + 1 < T) {
      bf16_t* nA = &sA[(buf ^ 1) * LBUF];
      bf16_t* nW = &sW[(buf ^ 1) * LBUF];
      *(v8bf*)&nA[sOff0] = ra0;
      *(v8bf*)&nA[sOff1] = ra1;
      *(v8bf*)&nW[sOff0] = rw0;
      *(v8bf*)&nW[sOff1] = rw1;
    }
    buf ^= 1;
  }

#pragma unroll
  for (int mi = 0; mi < 4; ++mi)
#pragma unroll
    for (int ni = 0; ni < 2; ++ni) {
      int col = nBlk + wn * 32 + ni * 16 + (lane & 15);
      int rowBase = mBlk + wm * 64 + mi * 16 + (lane >> 4) * 8;
      v8f a = acc[mi][ni];
#pragma unroll
      for (int e = 0; e < 8; ++e) {
        long long idx = cOff + (long long)(rowBase + e) * N + col;
        float val = a[e] * alpha;
        if (EPI == 1) val += R[idx];
        Cf[idx] = val;
      }
    }
}

// ---------------------------------------------------------------------------
// Fused gate+up GEMM: out_bf16 = silu(A@Wg^T) * (A@Wu^T)
// Block tile 128x64, waves 4(M)x2(N), wave tile 32x32 with dual accumulators.
// ---------------------------------------------------------------------------
constexpr int GBN = 64;

__global__ __launch_bounds__(256) void gemm_gateup_k(
    const bf16_t* __restrict__ A, const bf16_t* __restrict__ Wg,
    const bf16_t* __restrict__ Wu, bf16_t* __restrict__ Cb,
    int M, int N, int K) {
  __shared__ __align__(16) bf16_t sA[BM * LK];
  __shared__ __align__(16) bf16_t sG[GBN * LK];
  __shared__ __align__(16) bf16_t sU[GBN * LK];

  int tid = threadIdx.x;
  int lane = tid & 31, wid = tid >> 5;
  int wm = wid & 3, wn = wid >> 2;
  int mBlk = blockIdx.y * BM, nBlk = blockIdx.x * GBN;

  v8f ag[2][2] = {};
  v8f au[2][2] = {};

  int r0 = tid >> 2, cc0 = tid & 3;
  int r1 = r0 + 64;
  int rw = tid >> 2, ccw = tid & 3;

  for (int kt = 0; kt < K; kt += BK) {
    *(v8bf*)&sA[r0 * LK + cc0 * 8] =
        *(const v8bf*)&A[(long long)(mBlk + r0) * K + kt + cc0 * 8];
    *(v8bf*)&sA[r1 * LK + cc0 * 8] =
        *(const v8bf*)&A[(long long)(mBlk + r1) * K + kt + cc0 * 8];
    *(v8bf*)&sG[rw * LK + ccw * 8] =
        *(const v8bf*)&Wg[(long long)(nBlk + rw) * K + kt + ccw * 8];
    *(v8bf*)&sU[rw * LK + ccw * 8] =
        *(const v8bf*)&Wu[(long long)(nBlk + rw) * K + kt + ccw * 8];
    if (kt + BK < K) {
      __builtin_prefetch(&A[(long long)(mBlk + r0) * K + kt + BK + cc0 * 8], 0, 1);
      __builtin_prefetch(&Wg[(long long)(nBlk + rw) * K + kt + BK + ccw * 8], 0, 1);
    }
    __syncthreads();

    int rr = lane & 15, half = lane >> 4;
    int k0 = half * 8, k1 = 16 + half * 8;
    const bf16_t* pa0 = &sA[(wm * 32 + rr) * LK];
    const bf16_t* pa1 = &sA[(wm * 32 + 16 + rr) * LK];
    const bf16_t* pg0 = &sG[(wn * 32 + rr) * LK];
    const bf16_t* pg1 = &sG[(wn * 32 + 16 + rr) * LK];
    const bf16_t* pu0 = &sU[(wn * 32 + rr) * LK];
    const bf16_t* pu1 = &sU[(wn * 32 + 16 + rr) * LK];
    FragBF a0, a1, g0, g1, u0, u1;
    a0.h[0] = *(const v8bf*)(pa0 + k0); a0.h[1] = *(const v8bf*)(pa0 + k1);
    a1.h[0] = *(const v8bf*)(pa1 + k0); a1.h[1] = *(const v8bf*)(pa1 + k1);
    g0.h[0] = *(const v8bf*)(pg0 + k0); g0.h[1] = *(const v8bf*)(pg0 + k1);
    g1.h[0] = *(const v8bf*)(pg1 + k0); g1.h[1] = *(const v8bf*)(pg1 + k1);
    u0.h[0] = *(const v8bf*)(pu0 + k0); u0.h[1] = *(const v8bf*)(pu0 + k1);
    u1.h[0] = *(const v8bf*)(pu1 + k0); u1.h[1] = *(const v8bf*)(pu1 + k1);

    ag[0][0] = __builtin_amdgcn_wmma_f32_16x16x32_bf16(false, a0.v, false, g0.v, (short)0, ag[0][0], false, false);
    ag[0][1] = __builtin_amdgcn_wmma_f32_16x16x32_bf16(false, a0.v, false, g1.v, (short)0, ag[0][1], false, false);
    ag[1][0] = __builtin_amdgcn_wmma_f32_16x16x32_bf16(false, a1.v, false, g0.v, (short)0, ag[1][0], false, false);
    ag[1][1] = __builtin_amdgcn_wmma_f32_16x16x32_bf16(false, a1.v, false, g1.v, (short)0, ag[1][1], false, false);
    au[0][0] = __builtin_amdgcn_wmma_f32_16x16x32_bf16(false, a0.v, false, u0.v, (short)0, au[0][0], false, false);
    au[0][1] = __builtin_amdgcn_wmma_f32_16x16x32_bf16(false, a0.v, false, u1.v, (short)0, au[0][1], false, false);
    au[1][0] = __builtin_amdgcn_wmma_f32_16x16x32_bf16(false, a1.v, false, u0.v, (short)0, au[1][0], false, false);
    au[1][1] = __builtin_amdgcn_wmma_f32_16x16x32_bf16(false, a1.v, false, u1.v, (short)0, au[1][1], false, false);
    __syncthreads();
  }

#pragma unroll
  for (int mi = 0; mi < 2; ++mi)
#pragma unroll
    for (int ni = 0; ni < 2; ++ni) {
      int col = nBlk + wn * 32 + ni * 16 + (lane & 15);
      int rowBase = mBlk + wm * 32 + mi * 16 + (lane >> 4) * 8;
      v8f g = ag[mi][ni];
      v8f u = au[mi][ni];
#pragma unroll
      for (int e = 0; e < 8; ++e) {
        long long idx = (long long)(rowBase + e) * N + col;
        float gv = g[e];
        float sig = 1.0f / (1.0f + __expf(-gv));
        Cb[idx] = f2bf(gv * sig * u[e]);
      }
    }
}

// ---------------------------------------------------------------------------
// RoPE + pack: qf/kf fp32 [tok, heads*HD] -> head-major bf16; V transposed.
// ---------------------------------------------------------------------------
__global__ __launch_bounds__(256) void rope_pack_k(
    const float* __restrict__ qf, const float* __restrict__ kf,
    const float* __restrict__ vf, const int* __restrict__ pos,
    bf16_t* __restrict__ qb, bf16_t* __restrict__ kb, bf16_t* __restrict__ vtb) {
  int tok = blockIdx.x;              // 0..B*S-1
  int b = tok / S_, s = tok % S_;
  float p = (float)pos[tok];

  for (int e = threadIdx.x; e < NQ_ * HD_; e += blockDim.x) {
    int h = e >> 7, d = e & 127, j = d & 63;
    float inv = __expf(-((float)(2 * j) * (1.0f / HD_)) * LOG_THETA_);
    float ang = p * inv;
    float c = __cosf(ang), sn = __sinf(ang);
    float x0 = qf[(long long)tok * (NQ_ * HD_) + e];
    float xo = qf[(long long)tok * (NQ_ * HD_) + (h << 7) + ((d < 64) ? d + 64 : d - 64)];
    float o = (d < 64) ? (x0 * c - xo * sn) : (x0 * c + xo * sn);
    qb[((long long)(b * NQ_ + h) * S_ + s) * HD_ + d] = f2bf(o);
  }
  for (int e = threadIdx.x; e < NKV_ * HD_; e += blockDim.x) {
    int h = e >> 7, d = e & 127, j = d & 63;
    float inv = __expf(-((float)(2 * j) * (1.0f / HD_)) * LOG_THETA_);
    float ang = p * inv;
    float c = __cosf(ang), sn = __sinf(ang);
    float x0 = kf[(long long)tok * (NKV_ * HD_) + e];
    float xo = kf[(long long)tok * (NKV_ * HD_) + (h << 7) + ((d < 64) ? d + 64 : d - 64)];
    float o = (d < 64) ? (x0 * c - xo * sn) : (x0 * c + xo * sn);
    kb[((long long)(b * NKV_ + h) * S_ + s) * HD_ + d] = f2bf(o);
  }
  for (int e = threadIdx.x; e < NKV_ * HD_; e += blockDim.x) {
    int h = e >> 7, d = e & 127;
    vtb[((long long)(b * NKV_ + h) * HD_ + d) * S_ + s] =
        f2bf(vf[(long long)tok * (NKV_ * HD_) + e]);
  }
}

// ---------------------------------------------------------------------------
// Row softmax over S_=512 keys (no mask, matches reference), bf16 probs out.
// ---------------------------------------------------------------------------
__global__ __launch_bounds__(256) void softmax_k(const float* __restrict__ sc,
                                                 bf16_t* __restrict__ pr) {
  long long row = blockIdx.x;
  const float* sr = sc + row * S_;
  int tid = threadIdx.x;
  float a = sr[tid], b = sr[tid + 256];
  __shared__ float red[256];
  red[tid] = fmaxf(a, b);
  __syncthreads();
  for (int o = 128; o > 0; o >>= 1) {
    if (tid < o) red[tid] = fmaxf(red[tid], red[tid + o]);
    __syncthreads();
  }
  float mx = red[0];
  __syncthreads();
  float e0 = __expf(a - mx), e1 = __expf(b - mx);
  red[tid] = e0 + e1;
  __syncthreads();
  for (int o = 128; o > 0; o >>= 1) {
    if (tid < o) red[tid] += red[tid + o];
    __syncthreads();
  }
  float inv = 1.0f / red[0];
  pr[row * S_ + tid] = f2bf(e0 * inv);
  pr[row * S_ + tid + 256] = f2bf(e1 * inv);
}

// ---------------------------------------------------------------------------
// Gather attention output: [B,NQ,S,HD] fp32 -> [B*S, NQ*HD] bf16
// ---------------------------------------------------------------------------
__global__ __launch_bounds__(256) void gather_o_k(const float* __restrict__ of,
                                                  bf16_t* __restrict__ ob) {
  long long n = (long long)B_ * S_ * NQ_ * HD_;
  long long i = (long long)blockIdx.x * blockDim.x + threadIdx.x;
  long long st = (long long)gridDim.x * blockDim.x;
  for (; i < n; i += st) {
    int d = (int)(i & 127);
    int h = (int)((i >> 7) & 15);
    int s = (int)((i >> 11) & 511);
    int b = (int)(i >> 20);
    ob[i] = f2bf(of[(((long long)(b * NQ_ + h)) * S_ + s) * HD_ + d]);
  }
}

// ---------------------------------------------------------------------------
// Host orchestration
// ---------------------------------------------------------------------------
extern "C" void kernel_launch(void* const* d_in, const int* in_sizes, int n_in,
                              void* d_out, int out_size, void* d_ws, size_t ws_size,
                              hipStream_t stream) {
  const float* x   = (const float*)d_in[0];
  const int*   pos = (const int*)d_in[1];
  const float* wq  = (const float*)d_in[2];
  const float* wk  = (const float*)d_in[3];
  const float* wv  = (const float*)d_in[4];
  const float* wo  = (const float*)d_in[5];
  const float* wg  = (const float*)d_in[6];
  const float* wu  = (const float*)d_in[7];
  const float* wd  = (const float*)d_in[8];
  const float* ln1 = (const float*)d_in[9];
  const float* ln2 = (const float*)d_in[10];
  float* out = (float*)d_out;

  char* ws = (char*)d_ws;
  size_t o = 0;
  auto seg = [&](size_t bytes) -> char* {
    char* p = ws + o;
    o += (bytes + 255) & ~(size_t)255;
    return p;
  };

  const long long TOK = (long long)B_ * S_;      // 4096
  // bf16 weights
  bf16_t* wq_b = (bf16_t*)seg((size_t)NQ_ * HD_ * H_ * 2);
  bf16_t* wk_b = (bf16_t*)seg((size_t)NKV_ * HD_ * H_ * 2);
  bf16_t* wv_b = (bf16_t*)seg((size_t)NKV_ * HD_ * H_ * 2);
  bf16_t* wo_b = (bf16_t*)seg((size_t)H_ * NQ_ * HD_ * 2);
  bf16_t* wg_b = (bf16_t*)seg((size_t)INTER_ * H_ * 2);
  bf16_t* wu_b = (bf16_t*)seg((size_t)INTER_ * H_ * 2);
  bf16_t* wd_b = (bf16_t*)seg((size_t)H_ * INTER_ * 2);
  // activations
  bf16_t* h1  = (bf16_t*)seg((size_t)TOK * H_ * 2);
  float*  qf  = (float*)seg((size_t)TOK * NQ_ * HD_ * 4);   // reused later as ob
  float*  kf  = (float*)seg((size_t)TOK * NKV_ * HD_ * 4);  // kf+vf reused as h2
  float*  vf  = (float*)seg((size_t)TOK * NKV_ * HD_ * 4);
  bf16_t* qb  = (bf16_t*)seg((size_t)TOK * NQ_ * HD_ * 2);
  bf16_t* kb  = (bf16_t*)seg((size_t)TOK * NKV_ * HD_ * 2);
  bf16_t* vtb = (bf16_t*)seg((size_t)TOK * NKV_ * HD_ * 2);
  float*  scores = (float*)seg((size_t)B_ * NQ_ * S_ * S_ * 4); // reused as hmlp
  bf16_t* probs  = (bf16_t*)seg((size_t)B_ * NQ_ * S_ * S_ * 2);
  float*  of     = (float*)seg((size_t)B_ * NQ_ * S_ * HD_ * 4);
  float*  attn_x = (float*)seg((size_t)TOK * H_ * 4);
  // buffer reuse (lifetimes verified: producers run strictly after last reader)
  bf16_t* ob   = (bf16_t*)qf;      // 16.8MB into 33.5MB slot
  bf16_t* h2   = (bf16_t*)kf;      // 16.8MB into kf+vf (16.8MB)
  bf16_t* hmlp = (bf16_t*)scores;  // 67MB into 134MB slot

  // 1) weights -> bf16
  cvt_k<<<2048, 256, 0, stream>>>(wq, wq_b, (long long)NQ_ * HD_ * H_);
  cvt_k<<<2048, 256, 0, stream>>>(wk, wk_b, (long long)NKV_ * HD_ * H_);
  cvt_k<<<2048, 256, 0, stream>>>(wv, wv_b, (long long)NKV_ * HD_ * H_);
  cvt_k<<<2048, 256, 0, stream>>>(wo, wo_b, (long long)H_ * NQ_ * HD_);
  cvt_k<<<2048, 256, 0, stream>>>(wg, wg_b, (long long)INTER_ * H_);
  cvt_k<<<2048, 256, 0, stream>>>(wu, wu_b, (long long)INTER_ * H_);
  cvt_k<<<2048, 256, 0, stream>>>(wd, wd_b, (long long)H_ * INTER_);

  // 2) RMSNorm 1
  rmsnorm_k<<<(int)TOK, 256, 0, stream>>>(x, ln1, h1);

  // 3) Q/K/V projections (fp32 out, pre-RoPE)
  gemm_k<0><<<dim3((NQ_ * HD_) / BN, TOK / BM, 1), 256, 0, stream>>>(
      h1, wq_b, qf, nullptr, (int)TOK, NQ_ * HD_, H_, 0, 0, 0, 1, 1.0f);
  gemm_k<0><<<dim3((NKV_ * HD_) / BN, TOK / BM, 1), 256, 0, stream>>>(
      h1, wk_b, kf, nullptr, (int)TOK, NKV_ * HD_, H_, 0, 0, 0, 1, 1.0f);
  gemm_k<0><<<dim3((NKV_ * HD_) / BN, TOK / BM, 1), 256, 0, stream>>>(
      h1, wv_b, vf, nullptr, (int)TOK, NKV_ * HD_, H_, 0, 0, 0, 1, 1.0f);

  // 4) RoPE + head-major pack + V transpose
  rope_pack_k<<<(int)TOK, 256, 0, stream>>>(qf, kf, vf, pos, qb, kb, vtb);

  // 5) scores = SCALE * Q @ K^T  (batched over B*NQ; K/V shared per GQA group)
  gemm_k<0><<<dim3(S_ / BN, S_ / BM, B_ * NQ_), 256, 0, stream>>>(
      qb, kb, scores, nullptr, S_, S_, HD_,
      (long long)S_ * HD_, (long long)S_ * HD_, (long long)S_ * S_, G_, SCALE_);

  // 6) softmax -> bf16 probs
  softmax_k<<<B_ * NQ_ * S_, 256, 0, stream>>>(scores, probs);

  // 7) O = P @ V  (W = V^T, [HD, S] per kv head)
  gemm_k<0><<<dim3(HD_ / BN, S_ / BM, B_ * NQ_), 256, 0, stream>>>(
      probs, vtb, of, nullptr, S_, HD_, S_,
      (long long)S_ * S_, (long long)HD_ * S_, (long long)S_ * HD_, G_, 1.0f);

  // 8) [B,NQ,S,HD] -> [B*S, NQ*HD] bf16
  gather_o_k<<<2048, 256, 0, stream>>>(of, ob);

  // 9) attn_x = x + O @ Wo^T
  gemm_k<1><<<dim3(H_ / BN, TOK / BM, 1), 256, 0, stream>>>(
      ob, wo_b, attn_x, x, (int)TOK, H_, NQ_ * HD_, 0, 0, 0, 1, 1.0f);

  // 10) RMSNorm 2
  rmsnorm_k<<<(int)TOK, 256, 0, stream>>>(attn_x, ln2, h2);

  // 11) hmlp = silu(h2 @ Wg^T) * (h2 @ Wu^T)   (fused dual GEMM)
  gemm_gateup_k<<<dim3(INTER_ / GBN, TOK / BM, 1), 256, 0, stream>>>(
      h2, wg_b, wu_b, hmlp, (int)TOK, INTER_, H_);

  // 12) out = attn_x + hmlp @ Wd^T
  gemm_k<1><<<dim3(H_ / BN, TOK / BM, 1), 256, 0, stream>>>(
      hmlp, wd_b, out, attn_x, (int)TOK, H_, INTER_, 0, 0, 0, 1, 1.0f);

  (void)in_sizes; (void)n_in; (void)out_size; (void)ws_size;
}
